// LatentPixelAttention_50654844289166
// MI455X (gfx1250) — matrix operand order repaired and checked by reading, not compiled, not run on hardware
//
#include <hip/hip_runtime.h>

typedef __attribute__((ext_vector_type(16))) _Float16 v16h;
typedef __attribute__((ext_vector_type(8)))  _Float16 v8h;
typedef __attribute__((ext_vector_type(8)))  float    v8f;

union HV16 { v16h v; v8h h[2]; };

#define WMMA_F16(A, B, C) \
  __builtin_amdgcn_wmma_f32_16x16x32_f16(false, (A), false, (B), (short)0, (C), false, false)

// Problem constants
static constexpr int BATCH = 4;
static constexpr int C     = 64;
static constexpr int HW    = 4096;   // 64*64
static constexpr int QC    = 192;    // 3*C

// ---------------------------------------------------------------------------
// 1) 1x1 conv: qkv[b][oc][n] = sum_ic w[oc][ic] * x[b][ic][n]
// ---------------------------------------------------------------------------
__global__ __launch_bounds__(256) void conv1x1_kernel(const float* __restrict__ x,
                                                      const float* __restrict__ w,
                                                      float* __restrict__ qkv) {
  int idx = blockIdx.x * 256 + threadIdx.x;          // [0, 4*192*4096)
  int n  = idx & (HW - 1);
  int oc = (idx >> 12) % QC;
  int b  = idx / (QC * HW);
  const float* xb = x + (size_t)b * C * HW + n;
  const float* wr = w + oc * C;
  float s = 0.f;
#pragma unroll 8
  for (int ic = 0; ic < C; ++ic) s = fmaf(xb[(size_t)ic * HW], wr[ic], s);
  qkv[idx] = s;
}

// ---------------------------------------------------------------------------
// 2) depthwise 3x3 SAME conv, groups = 192
// ---------------------------------------------------------------------------
__global__ __launch_bounds__(256) void dwconv_kernel(const float* __restrict__ qkv,
                                                     const float* __restrict__ wdw,
                                                     float* __restrict__ out) {
  int idx = blockIdx.x * 256 + threadIdx.x;          // [0, 4*192*4096)
  int n  = idx & (HW - 1);
  int ch = (idx >> 12) % QC;
  int b  = idx / (QC * HW);
  int py = n >> 6, px = n & 63;
  const float* wp  = wdw + ch * 9;
  const float* src = qkv + (size_t)(b * QC + ch) * HW;
  float s = 0.f;
#pragma unroll
  for (int ky = 0; ky < 3; ++ky) {
    int y = py + ky - 1;
    if ((unsigned)y >= 64u) continue;
#pragma unroll
    for (int kx = 0; kx < 3; ++kx) {
      int xx = px + kx - 1;
      if ((unsigned)xx >= 64u) continue;
      s = fmaf(src[y * 64 + xx], wp[ky * 3 + kx], s);
    }
  }
  out[idx] = s;
}

// ---------------------------------------------------------------------------
// 3) per-(b,c) L2 norm of q and k spatial vectors -> inverse scales
// ---------------------------------------------------------------------------
__global__ __launch_bounds__(256) void norm_kernel(const float* __restrict__ qkvd,
                                                   float* __restrict__ qscale,
                                                   float* __restrict__ kscale) {
  int id = blockIdx.x;                 // [0, 512): c | b | t
  int c  = id & 63;
  int b  = (id >> 6) & 3;
  int t  = id >> 8;                    // 0 = q, 1 = k
  const float* src = qkvd + ((size_t)b * QC + t * C + c) * HW;
  float s = 0.f;
  for (int i = threadIdx.x; i < HW; i += 256) { float v = src[i]; s = fmaf(v, v, s); }
  __shared__ float red[256];
  red[threadIdx.x] = s;
  __syncthreads();
  for (int o = 128; o > 0; o >>= 1) {
    if (threadIdx.x < o) red[threadIdx.x] += red[threadIdx.x + o];
    __syncthreads();
  }
  if (threadIdx.x == 0) {
    float sc = 1.0f / fmaxf(sqrtf(red[0]), 1e-12f);
    (t ? kscale : qscale)[b * 64 + c] = sc;
  }
}

// ---------------------------------------------------------------------------
// 4) pack normalized q,k into f16 [b][n][c] layout (WMMA-friendly)
// ---------------------------------------------------------------------------
__global__ __launch_bounds__(256) void packqk_kernel(const float* __restrict__ qkvd,
                                                     const float* __restrict__ qscale,
                                                     const float* __restrict__ kscale,
                                                     _Float16* __restrict__ qh,
                                                     _Float16* __restrict__ kh) {
  int idx = blockIdx.x * 256 + threadIdx.x;          // [0, 4*64*4096)
  int n = idx & (HW - 1);
  int c = (idx >> 12) & 63;
  int b = idx >> 18;
  float qv = qkvd[((size_t)b * QC + c) * HW + n]       * qscale[b * 64 + c];
  float kv = qkvd[((size_t)b * QC + C + c) * HW + n]   * kscale[b * 64 + c];
  size_t o = ((size_t)b * HW + n) * 64 + c;
  qh[o] = (_Float16)qv;
  kh[o] = (_Float16)kv;
}

// ---------------------------------------------------------------------------
// 5) Pass A: per-row online softmax stats of S = Q^T K  (WMMA, f16->f32)
//    grid (32, 4), block 256 (8 waves); wave handles 16 rows i.
// ---------------------------------------------------------------------------
__global__ __launch_bounds__(256) void rowstats_kernel(const _Float16* __restrict__ qh,
                                                       const _Float16* __restrict__ kh,
                                                       float* __restrict__ rowmax,
                                                       float* __restrict__ rowsum) {
  const int b    = blockIdx.y;
  const int wave = threadIdx.x >> 5;
  const int lane = threadIdx.x & 31;
  const int l16  = lane & 15;
  const int hv   = lane >> 4;          // half of wave
  const int klo  = hv * 16;
  const int ibase = blockIdx.x * 128 + wave * 16;

  const _Float16* qb = qh + (size_t)b * HW * 64;
  const _Float16* kb = kh + (size_t)b * HW * 64;

  // A operands (Q rows, fixed for whole loop): c0 = 0 and c0 = 32
  HV16 a0, a1;
  const _Float16* qrow = qb + (size_t)(ibase + l16) * 64;
  a0.h[0] = *(const v8h*)(qrow + klo);
  a0.h[1] = *(const v8h*)(qrow + klo + 8);
  a1.h[0] = *(const v8h*)(qrow + 32 + klo);
  a1.h[1] = *(const v8h*)(qrow + 32 + klo + 8);

  float m[8], l[8];
#pragma unroll
  for (int v = 0; v < 8; ++v) { m[v] = -3.0e38f; l[v] = 0.f; }

  for (int j = 0; j < HW; j += 16) {
    const _Float16* krow = kb + (size_t)(j + l16) * 64 + hv * 16;
    __builtin_prefetch(krow + 16 * 64, 0, 1);        // next K tile
    HV16 b0, b1;
    b0.h[0] = *(const v8h*)(krow + 0);
    b0.h[1] = *(const v8h*)(krow + 8);
    b1.h[0] = *(const v8h*)(krow + 32);
    b1.h[1] = *(const v8h*)(krow + 40);

    v8f s = {};
    s = WMMA_F16(a0.v, b0.v, s);
    s = WMMA_F16(a1.v, b1.v, s);

#pragma unroll
    for (int v = 0; v < 8; ++v) {
      float sv = s[v];
      float mn = fmaxf(m[v], sv);
      l[v] = l[v] * __expf(m[v] - mn) + __expf(sv - mn);
      m[v] = mn;
    }
  }

  // combine (m,l) across the 16 lanes of each half (columns of the tile)
#pragma unroll
  for (int off = 1; off <= 8; off <<= 1) {
#pragma unroll
    for (int v = 0; v < 8; ++v) {
      float mo = __shfl_xor(m[v], off, 32);
      float lo = __shfl_xor(l[v], off, 32);
      float mn = fmaxf(m[v], mo);
      l[v] = l[v] * __expf(m[v] - mn) + lo * __expf(mo - mn);
      m[v] = mn;
    }
  }

  if (l16 == 0) {
    int r0 = b * HW + ibase + hv * 8;  // rows M = v + 8*hv
#pragma unroll
    for (int v = 0; v < 8; ++v) {
      rowmax[r0 + v] = m[v];
      rowsum[r0 + v] = fmaxf(l[v], 1e-30f);
    }
  }
}

// ---------------------------------------------------------------------------
// 6) pack V / rowsum into f16 [b][c][i] layout
// ---------------------------------------------------------------------------
__global__ __launch_bounds__(256) void packv_kernel(const float* __restrict__ qkvd,
                                                    const float* __restrict__ rowsum,
                                                    _Float16* __restrict__ vph) {
  int idx = blockIdx.x * 256 + threadIdx.x;          // [0, 4*64*4096)
  int n = idx & (HW - 1);
  int c = (idx >> 12) & 63;
  int b = idx >> 18;
  float vv = qkvd[((size_t)b * QC + 2 * C + c) * HW + n];
  float rs = rowsum[b * HW + n];
  vph[((size_t)b * C + c) * HW + n] = (_Float16)(vv / rs);
}

// ---------------------------------------------------------------------------
// 7) Pass B: out[c][n] = sum_i (v[c][i]/rowsum[i]) * exp(S[i][n]-rowmax[i])
//    grid (64, 4), block 256 (8 waves). Block = 64 c-rows x 64 n-cols.
//    Cooperative: per superstep the 8 waves jointly build E (128i x 64n) in
//    LDS (each wave computes a disjoint 32i x 32n sub-block -> no redundant
//    S/exp work), barrier, then each wave consumes E for its own 16c x 32n
//    output accumulation with a second WMMA.
// ---------------------------------------------------------------------------
__global__ __launch_bounds__(256) void attnout_kernel(const _Float16* __restrict__ qh,
                                                      const _Float16* __restrict__ kh,
                                                      const _Float16* __restrict__ vph,
                                                      const float* __restrict__ rowmax,
                                                      float* __restrict__ ao) {
  __shared__ float    smax[HW];            // 16 KB
  __shared__ _Float16 stage[4][64][32];    // 16 KB : [i_chunk][n_local][i_local]

  const int b = blockIdx.y;
  for (int t = threadIdx.x; t < HW; t += 256) smax[t] = rowmax[b * HW + t];
  __syncthreads();

  const int wave = threadIdx.x >> 5;
  const int lane = threadIdx.x & 31;
  const int l16  = lane & 15;
  const int hv   = lane >> 4;
  const int klo  = hv * 16;
  const int nblk = blockIdx.x * 64;        // block's first n column

  // writer role: this wave produces E for i-subchunk icW (32 i's) and
  // n-group ngW (32 n's) of the block tile
  const int icW = wave >> 1;               // 0..3
  const int ngW = wave & 1;                // 0..1
  // reader role: this wave accumulates c-tile ct (16 c's) x n-group np (32 n's)
  const int ct = wave & 3;                 // 0..3
  const int np = wave >> 2;                // 0..1
  const int cbase = ct * 16;

  const _Float16* qb = qh  + (size_t)b * HW * 64;
  const _Float16* kb = kh  + (size_t)b * HW * 64;
  const _Float16* vb = vph + (size_t)b * C * HW;

  // Preload writer's K B-operands (its two 16-col n-subtiles; loop-invariant)
  HV16 kop[2][2];
#pragma unroll
  for (int s = 0; s < 2; ++s) {
    const _Float16* kr = kb + (size_t)(nblk + ngW * 32 + s * 16 + l16) * 64 + hv * 16;
    kop[s][0].h[0] = *(const v8h*)(kr + 0);
    kop[s][0].h[1] = *(const v8h*)(kr + 8);
    kop[s][1].h[0] = *(const v8h*)(kr + 32);
    kop[s][1].h[1] = *(const v8h*)(kr + 40);
  }

  v8f acc[2];
  acc[0] = (v8f){};
  acc[1] = (v8f){};

  for (int i0 = 0; i0 < HW; i0 += 128) {
    // ---------------- writer phase: build E(128i x 64n) in LDS ------------
    const int ibaseW = i0 + icW * 32;
    HV16 aq[2][2];
    float mx[2][8];
#pragma unroll
    for (int it = 0; it < 2; ++it) {
      const _Float16* qr = qb + (size_t)(ibaseW + it * 16 + l16) * 64;
      aq[it][0].h[0] = *(const v8h*)(qr + klo);
      aq[it][0].h[1] = *(const v8h*)(qr + klo + 8);
      aq[it][1].h[0] = *(const v8h*)(qr + 32 + klo);
      aq[it][1].h[1] = *(const v8h*)(qr + 32 + klo + 8);
#pragma unroll
      for (int v = 0; v < 8; ++v) mx[it][v] = smax[ibaseW + it * 16 + hv * 8 + v];
    }
#pragma unroll
    for (int s = 0; s < 2; ++s) {
#pragma unroll
      for (int it = 0; it < 2; ++it) {
        v8f sv = {};
        sv = WMMA_F16(aq[it][0].v, kop[s][0].v, sv);
        sv = WMMA_F16(aq[it][1].v, kop[s][1].v, sv);
        int rbase = it * 16 + hv * 8;    // i_local within 32-chunk
        int ncol  = ngW * 32 + s * 16 + l16;
#pragma unroll
        for (int v = 0; v < 8; v += 2) {
          float e0 = __expf(sv[v]     - mx[it][v]);
          float e1 = __expf(sv[v + 1] - mx[it][v + 1]);
          union { _Float16 h[2]; unsigned int u; } pk;
          pk.h[0] = (_Float16)e0;
          pk.h[1] = (_Float16)e1;
          *(unsigned int*)&stage[icW][ncol][rbase + v] = pk.u;
        }
      }
    }
    __syncthreads();

    // ---------------- reader phase: out += Vp * E -------------------------
#pragma unroll
    for (int ic = 0; ic < 4; ++ic) {
      HV16 av;
      const _Float16* vr = vb + (size_t)(cbase + l16) * HW + i0 + ic * 32;
      av.h[0] = *(const v8h*)(vr + klo);
      av.h[1] = *(const v8h*)(vr + klo + 8);
#pragma unroll
      for (int s = 0; s < 2; ++s) {
        HV16 be;
        const v8h* sp = (const v8h*)&stage[ic][np * 32 + s * 16 + l16][hv * 16];
        be.h[0] = sp[0];
        be.h[1] = sp[1];
        acc[s] = WMMA_F16(av.v, be.v, acc[s]);
      }
    }
    __syncthreads();
  }

  // Write out: row c = cbase + hv*8 + v, col n = nblk + np*32 + s*16 + l16
#pragma unroll
  for (int s = 0; s < 2; ++s) {
#pragma unroll
    for (int v = 0; v < 8; ++v) {
      ao[((size_t)(b * C + cbase + hv * 8 + v)) * HW + nblk + np * 32 + s * 16 + l16] = acc[s][v];
    }
  }
}

// ---------------------------------------------------------------------------
// 8) 3x3 SAME projection conv: d_out = conv(ao, w_proj)
//    grid (16, 64, 4): (row-group, oc, b), block 256 (4 rows x 64 cols)
// ---------------------------------------------------------------------------
__global__ __launch_bounds__(256) void proj_kernel(const float* __restrict__ ao,
                                                   const float* __restrict__ wproj,
                                                   float* __restrict__ out) {
  __shared__ float wl[C * 9];
  int oc = blockIdx.y, b = blockIdx.z;
  for (int t = threadIdx.x; t < C * 9; t += 256) wl[t] = wproj[oc * C * 9 + t];
  __syncthreads();
  int py = blockIdx.x * 4 + (threadIdx.x >> 6);
  int px = threadIdx.x & 63;
  const float* src = ao + (size_t)b * C * HW;
  float s = 0.f;
  for (int ic = 0; ic < C; ++ic) {
    const float* sp = src + (size_t)ic * HW;
    const float* wp = wl + ic * 9;
#pragma unroll
    for (int ky = 0; ky < 3; ++ky) {
      int y = py + ky - 1;
      if ((unsigned)y >= 64u) continue;
#pragma unroll
      for (int kx = 0; kx < 3; ++kx) {
        int xx = px + kx - 1;
        if ((unsigned)xx >= 64u) continue;
        s = fmaf(sp[y * 64 + xx], wp[ky * 3 + kx], s);
      }
    }
  }
  out[((size_t)(b * C + oc)) * HW + py * 64 + px] = s;
}

// ---------------------------------------------------------------------------
extern "C" void kernel_launch(void* const* d_in, const int* in_sizes, int n_in,
                              void* d_out, int out_size, void* d_ws, size_t ws_size,
                              hipStream_t stream) {
  (void)in_sizes; (void)n_in; (void)out_size; (void)ws_size;
  const float* x      = (const float*)d_in[0];
  const float* w_qkv  = (const float*)d_in[1];
  const float* w_dw   = (const float*)d_in[2];
  const float* w_proj = (const float*)d_in[3];
  float* out = (float*)d_out;
  char*  ws  = (char*)d_ws;

  // Workspace layout (256B aligned offsets)
  float*    qkv    = (float*)   (ws + 0);          // 4*192*4096 f32 = 12,582,912
  float*    qkvd   = (float*)   (ws + 12582912);   // 12,582,912
  _Float16* qh     = (_Float16*)(ws + 25165824);   // 4*4096*64 f16 = 2,097,152
  _Float16* kh     = (_Float16*)(ws + 27262976);   // 2,097,152
  _Float16* vph    = (_Float16*)(ws + 29360128);   // 2,097,152
  float*    qscale = (float*)   (ws + 31457280);   // 1,024
  float*    kscale = (float*)   (ws + 31458304);   // 1,024
  float*    rowmax = (float*)   (ws + 31459328);   // 65,536
  float*    rowsum = (float*)   (ws + 31524864);   // 65,536
  float*    ao     = (float*)   (ws + 31590400);   // 4,194,304   (end ~34.1 MB)

  conv1x1_kernel <<<dim3(BATCH * QC * HW / 256), 256, 0, stream>>>(x, w_qkv, qkv);
  dwconv_kernel  <<<dim3(BATCH * QC * HW / 256), 256, 0, stream>>>(qkv, w_dw, qkvd);
  norm_kernel    <<<dim3(512),                   256, 0, stream>>>(qkvd, qscale, kscale);
  packqk_kernel  <<<dim3(BATCH * C * HW / 256),  256, 0, stream>>>(qkvd, qscale, kscale, qh, kh);
  rowstats_kernel<<<dim3(32, BATCH),             256, 0, stream>>>(qh, kh, rowmax, rowsum);
  packv_kernel   <<<dim3(BATCH * C * HW / 256),  256, 0, stream>>>(qkvd, rowsum, vph);
  attnout_kernel <<<dim3(64, BATCH),             256, 0, stream>>>(qh, kh, vph, rowmax, ao);
  proj_kernel    <<<dim3(16, C, BATCH),          256, 0, stream>>>(ao, w_proj, out);
}
